// ThreeDSGAT_41558103556382
// MI455X (gfx1250) — compile-verified, matrix operand order checked by reading, not run on hardware
//
#include <hip/hip_runtime.h>
#include <hip/hip_bf16.h>

typedef __attribute__((ext_vector_type(16))) _Float16 v16h;
typedef __attribute__((ext_vector_type(8)))  _Float16 v8h;
typedef __attribute__((ext_vector_type(8)))  float    v8f;

#define BATCH 4
#define NNODE 4096
#define BN    16384      // BATCH * NNODE
#define CDIM  768
#define NH    12
#define HD    64
#define KNBR  6
#define QKVC  2304       // 3 * CDIM

// ---------------------------------------------------------------------------
// Fragment loader implementing the CDNA5 16-bit A/B WMMA VGPR layout
// (CDNA5 ISA §7.12.2):
//   lanes 0-15  : row = lane,    K = {k0..k0+7} in v[0..3], {k0+16..k0+23} in v[4..7]
//   lanes 16-31 : row = lane-16, same but K offset +8
// Both halves are 8 consecutive f16 = one global_load_b128 each; with the
// k-loop fully unrolled these fold to base + constant IOFFSET.
// ---------------------------------------------------------------------------
union Frag16 { v16h v; v8h h[2]; };

__device__ __forceinline__ v16h load_frag(const _Float16* __restrict__ p, int k0) {
  Frag16 f;
  f.h[0] = *reinterpret_cast<const v8h*>(p + k0);
  f.h[1] = *reinterpret_cast<const v8h*>(p + k0 + 16);
  return f.v;
}

#define WMMA_F16(A, B, Cacc) \
  __builtin_amdgcn_wmma_f32_16x16x32_f16(false, (A), false, (B), (short)0, (Cacc), false, false)

// ---------------------------------------------------------------------------
// f32 -> f16 conversion (grid-stride)
// ---------------------------------------------------------------------------
__global__ void cvt_f16_kernel(const float* __restrict__ src,
                               _Float16* __restrict__ dst, int n) {
  int i = blockIdx.x * blockDim.x + threadIdx.x;
  int stride = gridDim.x * blockDim.x;
  for (; i < n; i += stride) dst[i] = (_Float16)src[i];
}

// ---------------------------------------------------------------------------
// Shared GEMM core: wave computes a 32x64 tile, k fully unrolled (24 steps,
// 8 WMMAs each). All loads are constant-offset; scheduler hoists freely.
// ---------------------------------------------------------------------------
__device__ __forceinline__ void gemm_core_32x64(
    const _Float16* __restrict__ pa0, const _Float16* __restrict__ pa1,
    const _Float16* __restrict__ pb0, const _Float16* __restrict__ pb1,
    const _Float16* __restrict__ pb2, const _Float16* __restrict__ pb3,
    int koff, v8f acc[8]) {
#pragma unroll
  for (int kt = 0; kt < CDIM; kt += 32) {
    const int k0 = kt + koff;
    v16h a0 = load_frag(pa0, k0);
    v16h a1 = load_frag(pa1, k0);
    v16h b0 = load_frag(pb0, k0);
    v16h b1 = load_frag(pb1, k0);
    v16h b2 = load_frag(pb2, k0);
    v16h b3 = load_frag(pb3, k0);
    acc[0] = WMMA_F16(a0, b0, acc[0]);
    acc[1] = WMMA_F16(a0, b1, acc[1]);
    acc[2] = WMMA_F16(a0, b2, acc[2]);
    acc[3] = WMMA_F16(a0, b3, acc[3]);
    acc[4] = WMMA_F16(a1, b0, acc[4]);
    acc[5] = WMMA_F16(a1, b1, acc[5]);
    acc[6] = WMMA_F16(a1, b2, acc[6]);
    acc[7] = WMMA_F16(a1, b3, acc[7]);
  }
}

// ---------------------------------------------------------------------------
// QKV GEMM:  Y[16384 x 2304] = X[16384 x 768] @ Wqkv[2304 x 768]^T
// ---------------------------------------------------------------------------
__global__ void __launch_bounds__(128)
gemm_qkv_kernel(const _Float16* __restrict__ X, const _Float16* __restrict__ W,
                _Float16* __restrict__ Y) {
  const int lane  = threadIdx.x & 31;
  const int wave  = threadIdx.x >> 5;
  const int r16   = lane & 15;
  const int koff  = (lane >> 4) << 3;            // 0 or 8
  const int mBase = blockIdx.x * 32;
  const int nBase = blockIdx.y * 256 + wave * 64;

  const _Float16* pa0 = X + (size_t)(mBase + r16) * CDIM;
  const _Float16* pa1 = pa0 + (size_t)16 * CDIM;
  const _Float16* pb0 = W + (size_t)(nBase +  0 + r16) * CDIM;
  const _Float16* pb1 = W + (size_t)(nBase + 16 + r16) * CDIM;
  const _Float16* pb2 = W + (size_t)(nBase + 32 + r16) * CDIM;
  const _Float16* pb3 = W + (size_t)(nBase + 48 + r16) * CDIM;

  v8f acc[8] = {v8f{}, v8f{}, v8f{}, v8f{}, v8f{}, v8f{}, v8f{}, v8f{}};
  gemm_core_32x64(pa0, pa1, pb0, pb1, pb2, pb3, koff, acc);

  // C/D layout: lane 0-15 -> M = r (VGPR r), lane 16-31 -> M = r+8; N = lane&15
  const int rowHi = (lane >> 4) << 3;
#pragma unroll
  for (int r = 0; r < 8; ++r) {
    const size_t orow0 = (size_t)(mBase + rowHi + r) * QKVC;
    const size_t orow1 = orow0 + (size_t)16 * QKVC;
#pragma unroll
    for (int t = 0; t < 4; ++t) {
      Y[orow0 + nBase + t * 16 + r16] = (_Float16)acc[t][r];
      Y[orow1 + nBase + t * 16 + r16] = (_Float16)acc[4 + t][r];
    }
  }
}

// ---------------------------------------------------------------------------
// Neighbor attention: one wave per (node, head). hd=64 -> 2 elems per lane.
// ---------------------------------------------------------------------------
__global__ void __launch_bounds__(128)
attn_kernel(const _Float16* __restrict__ QKV, const int* __restrict__ nidx,
            const float* __restrict__ ndist, _Float16* __restrict__ AGG) {
  const int lane = threadIdx.x & 31;
  const int wave = threadIdx.x >> 5;
  const int id   = blockIdx.x * 4 + wave;        // [0, BN*NH)
  const int bn   = id / NH;
  const int head = id - bn * NH;
  const int b    = bn >> 12;
  const int node = bn & (NNODE - 1);
  const int d0   = lane * 2;

  const _Float16* qp = QKV + (size_t)bn * QKVC + head * HD;
  const float q0 = (float)qp[d0], q1 = (float)qp[d0 + 1];

  float sc[KNBR];
  int   jrow[KNBR];
#pragma unroll
  for (int kk = 0; kk < KNBR; ++kk) {
    const int j = nidx[node * KNBR + kk];
    jrow[kk] = (b << 12) + j;
    const _Float16* kp = QKV + (size_t)jrow[kk] * QKVC + CDIM + head * HD;
    float p = q0 * (float)kp[d0] + q1 * (float)kp[d0 + 1];
#pragma unroll
    for (int off = 16; off > 0; off >>= 1) p += __shfl_xor(p, off, 32);
    sc[kk] = p * 0.125f - ndist[node * KNBR + kk];   // scale=1/sqrt(64), gamma=1
  }

  float m = sc[0];
#pragma unroll
  for (int kk = 1; kk < KNBR; ++kk) m = fmaxf(m, sc[kk]);
  float w[KNBR], s = 0.0f;
#pragma unroll
  for (int kk = 0; kk < KNBR; ++kk) { w[kk] = __expf(sc[kk] - m); s += w[kk]; }
  const float inv = 1.0f / s;

  float a0 = 0.0f, a1 = 0.0f;
#pragma unroll
  for (int kk = 0; kk < KNBR; ++kk) {
    const _Float16* vp = QKV + (size_t)jrow[kk] * QKVC + 2 * CDIM + head * HD;
    const float wk = w[kk] * inv;
    a0 += wk * (float)vp[d0];
    a1 += wk * (float)vp[d0 + 1];
  }

  _Float16* op = AGG + (size_t)bn * CDIM + head * HD;
  op[d0]     = (_Float16)a0;
  op[d0 + 1] = (_Float16)a1;
}

// ---------------------------------------------------------------------------
// Output projection GEMM with fused bias + residual:
//   out[16384 x 768] = AGG @ Wo^T + bias + x      (f32 result)
// ---------------------------------------------------------------------------
__global__ void __launch_bounds__(128)
gemm_out_kernel(const _Float16* __restrict__ A, const _Float16* __restrict__ W,
                const float* __restrict__ bias, const float* __restrict__ x,
                float* __restrict__ out) {
  const int lane  = threadIdx.x & 31;
  const int wave  = threadIdx.x >> 5;
  const int r16   = lane & 15;
  const int koff  = (lane >> 4) << 3;
  const int mBase = blockIdx.x * 32;
  const int nBase = blockIdx.y * 256 + wave * 64;

  const _Float16* pa0 = A + (size_t)(mBase + r16) * CDIM;
  const _Float16* pa1 = pa0 + (size_t)16 * CDIM;
  const _Float16* pb0 = W + (size_t)(nBase +  0 + r16) * CDIM;
  const _Float16* pb1 = W + (size_t)(nBase + 16 + r16) * CDIM;
  const _Float16* pb2 = W + (size_t)(nBase + 32 + r16) * CDIM;
  const _Float16* pb3 = W + (size_t)(nBase + 48 + r16) * CDIM;

  v8f acc[8] = {v8f{}, v8f{}, v8f{}, v8f{}, v8f{}, v8f{}, v8f{}, v8f{}};
  gemm_core_32x64(pa0, pa1, pb0, pb1, pb2, pb3, koff, acc);

  const int rowHi = (lane >> 4) << 3;
#pragma unroll
  for (int r = 0; r < 8; ++r) {
    const int row0 = mBase + rowHi + r;
    const int row1 = row0 + 16;
    const size_t base0 = (size_t)row0 * CDIM;
    const size_t base1 = (size_t)row1 * CDIM;
#pragma unroll
    for (int t = 0; t < 4; ++t) {
      const int col = nBase + t * 16 + r16;
      out[base0 + col] = acc[t][r]     + bias[col] + x[base0 + col];
      out[base1 + col] = acc[4 + t][r] + bias[col] + x[base1 + col];
    }
  }
}

// ---------------------------------------------------------------------------
// In-place LayerNorm over C=768, one block (256 thr) per row.
// ---------------------------------------------------------------------------
__global__ void __launch_bounds__(256)
ln_kernel(float* __restrict__ out, const float* __restrict__ g,
          const float* __restrict__ bt) {
  __shared__ float s1[256];
  __shared__ float s2[256];
  const int row = blockIdx.x;
  const int tid = threadIdx.x;
  const size_t base = (size_t)row * CDIM;

  const float v0 = out[base + tid];
  const float v1 = out[base + tid + 256];
  const float v2 = out[base + tid + 512];
  s1[tid] = v0 + v1 + v2;
  s2[tid] = v0 * v0 + v1 * v1 + v2 * v2;
  __syncthreads();
  for (int off = 128; off > 0; off >>= 1) {
    if (tid < off) { s1[tid] += s1[tid + off]; s2[tid] += s2[tid + off]; }
    __syncthreads();
  }
  const float mu   = s1[0] * (1.0f / CDIM);
  const float var  = s2[0] * (1.0f / CDIM) - mu * mu;
  const float rstd = rsqrtf(var + 1e-5f);

  out[base + tid]       = (v0 - mu) * rstd * g[tid]       + bt[tid];
  out[base + tid + 256] = (v1 - mu) * rstd * g[tid + 256] + bt[tid + 256];
  out[base + tid + 512] = (v2 - mu) * rstd * g[tid + 512] + bt[tid + 512];
}

// ---------------------------------------------------------------------------
// Workspace layout (bytes):
//   X16   @ 0          : 16384*768  f16 = 25,165,824
//   Wqkv  @ 25165824   : 2304*768   f16 =  3,538,944
//   Wo16  @ 28704768   : 768*768    f16 =  1,179,648
//   QKV   @ 29884416   : 16384*2304 f16 = 75,497,472
//   AGG   @ 105381888  : 16384*768  f16 = 25,165,824   (total ~130.5 MB)
// ---------------------------------------------------------------------------
extern "C" void kernel_launch(void* const* d_in, const int* in_sizes, int n_in,
                              void* d_out, int out_size, void* d_ws, size_t ws_size,
                              hipStream_t stream) {
  const float* x    = (const float*)d_in[0];
  const float* Wq   = (const float*)d_in[1];
  const float* Wk   = (const float*)d_in[2];
  const float* Wv   = (const float*)d_in[3];
  const float* Wo   = (const float*)d_in[4];
  const float* Wob  = (const float*)d_in[5];
  const float* lng  = (const float*)d_in[6];
  const float* lnb  = (const float*)d_in[7];
  const int*   nidx = (const int*)d_in[8];
  const float* ndst = (const float*)d_in[9];
  float* out = (float*)d_out;

  char* ws = (char*)d_ws;
  _Float16* X16  = (_Float16*)(ws);
  _Float16* Wqkv = (_Float16*)(ws + (size_t)25165824);
  _Float16* Wo16 = (_Float16*)(ws + (size_t)28704768);
  _Float16* QKV  = (_Float16*)(ws + (size_t)29884416);
  _Float16* AGG  = (_Float16*)(ws + (size_t)105381888);

  // 1) convert activations + weights to f16 (Wq/Wk/Wv fused into one matrix)
  cvt_f16_kernel<<<2048, 256, 0, stream>>>(x,  X16,  BN * CDIM);
  cvt_f16_kernel<<<512,  256, 0, stream>>>(Wq, Wqkv,                   CDIM * CDIM);
  cvt_f16_kernel<<<512,  256, 0, stream>>>(Wk, Wqkv + CDIM * CDIM,     CDIM * CDIM);
  cvt_f16_kernel<<<512,  256, 0, stream>>>(Wv, Wqkv + 2 * CDIM * CDIM, CDIM * CDIM);
  cvt_f16_kernel<<<512,  256, 0, stream>>>(Wo, Wo16, CDIM * CDIM);

  // 2) fused QKV projection GEMM (project once per node, gather later)
  gemm_qkv_kernel<<<dim3(BN / 32, QKVC / 256), 128, 0, stream>>>(X16, Wqkv, QKV);

  // 3) neighbor attention (gather + softmax over K=6)
  attn_kernel<<<(BN * NH) / 4, 128, 0, stream>>>(QKV, nidx, ndst, AGG);

  // 4) output projection + bias + residual -> d_out (f32)
  gemm_out_kernel<<<dim3(BN / 32, CDIM / 256), 128, 0, stream>>>(AGG, Wo16, Wob, x, out);

  // 5) in-place LayerNorm
  ln_kernel<<<BN, 256, 0, stream>>>(out, lng, lnb);
}
